// SelfAttention_33578054320492
// MI455X (gfx1250) — matrix operand order repaired and checked by reading, not compiled, not run on hardware
//
#include <hip/hip_runtime.h>
#include <hip/hip_bf16.h>

#define B_      4
#define L_      2048
#define EMBED_  1024
#define HEADS_  16
#define HD_     64
#define M_      (B_ * L_)   // 8192

typedef __attribute__((ext_vector_type(16))) __bf16 v16bf;
typedef __attribute__((ext_vector_type(8)))  float  v8f;

union Frag16 { v16bf v; uint4 u[2]; };

// ---------------------------------------------------------------------------
// CDNA5 async global->LDS 16-byte copy (ASYNCcnt-tracked) + wait
// ---------------------------------------------------------------------------
__device__ __forceinline__ void async_ld16(void* lds, const void* g) {
#if defined(__HIP_DEVICE_COMPILE__)
    unsigned           l  = (unsigned)(unsigned long long)lds;   // as(3) offset
    unsigned long long ga = (unsigned long long)g;
    asm volatile("global_load_async_to_lds_b128 %0, %1, off"
                 :: "v"(l), "v"(ga) : "memory");
#endif
}

__device__ __forceinline__ void wait_async0() {
#if defined(__HIP_DEVICE_COMPILE__)
#if __has_builtin(__builtin_amdgcn_s_wait_asynccnt)
    __builtin_amdgcn_s_wait_asynccnt(0);
#else
    asm volatile("s_wait_asynccnt 0x0" ::: "memory");
#endif
#endif
}

// ---------------------------------------------------------------------------
// f32 -> bf16 conversion (x)
// ---------------------------------------------------------------------------
__global__ __launch_bounds__(256) void cvt_f32_bf16(const float* __restrict__ in,
                                                    __bf16* __restrict__ out, int n) {
    int i = blockIdx.x * blockDim.x + threadIdx.x;
    if (i < n) out[i] = (__bf16)in[i];
}

// ---------------------------------------------------------------------------
// Weight transpose + convert: Wt[n][k] (bf16) = W[k][n] (f32), 1024x1024.
// One-time cost; makes all GEMM B-staging contiguous 16B async copies.
// ---------------------------------------------------------------------------
__global__ __launch_bounds__(256) void cvt_w_transpose(const float* __restrict__ in,
                                                       __bf16* __restrict__ out) {
    int i = blockIdx.x * 256 + threadIdx.x;          // i = n*1024 + k
    int n = i >> 10, k = i & 1023;
    out[i] = (__bf16)in[(size_t)k * EMBED_ + n];
}

// ---------------------------------------------------------------------------
// GEMM: C[M x 1024] = A[M x 1024](bf16) @ W[1024 x 1024] + bias,
//       with W given transposed as Wt[N][K] (bf16).
// Workgroup tile 128x128, 8 waves each computing 32x64 (8 WMMA / k-step).
// out_mode 0: bf16 scatter into [B,H,L,D]       (Q, K)
// out_mode 1: f32 row-major [M x 1024]          (final output)
// out_mode 2: bf16 scatter into [B,H,D,L]       (V, pre-transposed for P@V)
// ---------------------------------------------------------------------------
__global__ __launch_bounds__(256) void gemm_bf16(const __bf16* __restrict__ A,
                                                 const __bf16* __restrict__ Wt,
                                                 const float* __restrict__ bias,
                                                 void* __restrict__ out, int out_mode) {
    const int K = EMBED_, N = EMBED_;
    __shared__ __align__(16) __bf16 Alds[128 * 32];   // [m][k]
    __shared__ __align__(16) __bf16 Blds[128 * 32];   // [n][k]

    const int tid  = threadIdx.x;
    const int lane = tid & 31;
    const int wave = tid >> 5;
    const int m_base = blockIdx.x * 128;
    const int n_base = blockIdx.y * 128;

    const int wm = wave & 3;        // 4 m-slices of 32 rows
    const int wn = wave >> 2;       // 2 n-slices of 64 cols
    const int row16 = lane & 15;
    const int hi    = lane >> 4;    // 0/1
    const int kb    = hi << 3;      // A/B frag K base within 32: 0 or 8
    const int kg    = hi << 3;      // B frag K base within 32: 0 or 8

    v8f acc[2][4];
    #pragma unroll
    for (int i = 0; i < 2; ++i)
        #pragma unroll
        for (int j = 0; j < 4; ++j)
            #pragma unroll
            for (int e = 0; e < 8; ++e) acc[i][j][e] = 0.0f;

    for (int k0 = 0; k0 < K; k0 += 32) {
        // Async-stage A tile 128x32 and B tile 128x32 (contiguous 16B chunks,
        // 512 chunks each, 2 per thread per tile) straight into LDS.
        {
            int c0 = tid, c1 = tid + 256;
            int r0 = c0 >> 2, kc0 = (c0 & 3) << 3;
            int r1 = c1 >> 2, kc1 = (c1 & 3) << 3;
            async_ld16(&Alds[r0 * 32 + kc0], &A[(size_t)(m_base + r0) * K + k0 + kc0]);
            async_ld16(&Alds[r1 * 32 + kc1], &A[(size_t)(m_base + r1) * K + k0 + kc1]);
            async_ld16(&Blds[r0 * 32 + kc0], &Wt[(size_t)(n_base + r0) * K + k0 + kc0]);
            async_ld16(&Blds[r1 * 32 + kc1], &Wt[(size_t)(n_base + r1) * K + k0 + kc1]);
        }
        wait_async0();
        __syncthreads();

        Frag16 af[2], bf[4];
        #pragma unroll
        for (int i = 0; i < 2; ++i) {
            const int arow = wm * 32 + i * 16 + row16;
            af[i].u[0] = *(const uint4*)&Alds[arow * 32 + kb];
            af[i].u[1] = *(const uint4*)&Alds[arow * 32 + kb + 16];
        }
        #pragma unroll
        for (int j = 0; j < 4; ++j) {
            const int bn = wn * 64 + j * 16 + row16;
            bf[j].u[0] = *(const uint4*)&Blds[bn * 32 + kg];
            bf[j].u[1] = *(const uint4*)&Blds[bn * 32 + kg + 16];
        }
        #pragma unroll
        for (int i = 0; i < 2; ++i)
            #pragma unroll
            for (int j = 0; j < 4; ++j)
                acc[i][j] = __builtin_amdgcn_wmma_f32_16x16x32_bf16(
                    false, af[i].v, false, bf[j].v, (short)0, acc[i][j], false, false);
        __syncthreads();
    }

    // Epilogue
    #pragma unroll
    for (int i = 0; i < 2; ++i) {
        #pragma unroll
        for (int j = 0; j < 4; ++j) {
            #pragma unroll
            for (int r = 0; r < 8; ++r) {
                const int m = m_base + wm * 32 + i * 16 + r + hi * 8;
                const int c = n_base + wn * 64 + j * 16 + row16;
                const float val = acc[i][j][r] + bias[c];
                if (out_mode == 0) {
                    const int b = m >> 11, l = m & 2047;
                    const int h = c >> 6,  d = c & 63;
                    ((__bf16*)out)[(((size_t)(b * HEADS_ + h) * L_ + l) << 6) + d] = (__bf16)val;
                } else if (out_mode == 2) {
                    const int b = m >> 11, l = m & 2047;
                    const int h = c >> 6,  d = c & 63;
                    ((__bf16*)out)[(((size_t)(b * HEADS_ + h) * HD_ + d) << 11) + l] = (__bf16)val;
                } else {
                    ((float*)out)[(size_t)m * N + c] = val;
                }
            }
        }
    }
}

// ---------------------------------------------------------------------------
// Flash attention: one block per (b, h, 128-query tile); 8 waves x 16 q-rows.
// Q,K: bf16 [B,H,L,64].  Vt: bf16 [B,H,64,L] (pre-transposed).
// O: bf16 [B*L, 1024] row-major (pre-Wo).
// ---------------------------------------------------------------------------
__global__ __launch_bounds__(256) void attn_kernel(const __bf16* __restrict__ Q,
                                                   const __bf16* __restrict__ Km,
                                                   const __bf16* __restrict__ Vt,
                                                   const int* __restrict__ mask,
                                                   __bf16* __restrict__ O) {
    __shared__ __align__(16) __bf16 Klds[64 * 64];       // [key][d]
    __shared__ __align__(16) __bf16 Vlds[64 * 64];       // [d][key]
    __shared__ __align__(16) __bf16 Plds[8][16 * 64];    // per-wave P tile

    const int tid  = threadIdx.x;
    const int lane = tid & 31;
    const int wave = tid >> 5;
    const int idx  = blockIdx.x;
    const int qblk = idx & 15;
    const int h    = (idx >> 4) & 15;
    const int b    = idx >> 8;

    const size_t head  = (size_t)(b * HEADS_ + h) * L_ * HD_;   // Q/K base
    const size_t vhead = head;                                   // same extent, [D,L] layout
    const int row16 = lane & 15;
    const int hi    = lane >> 4;
    const int kb    = hi << 3;
    const int q_base = qblk * 128 + wave * 16;

    // Q fragments for this wave's 16 rows (K=64 -> 2 fragments)
    Frag16 qf[2];
    {
        const __bf16* qp = Q + head + (size_t)(q_base + row16) * HD_;
        #pragma unroll
        for (int s = 0; s < 2; ++s) {
            qf[s].u[0] = *(const uint4*)(qp + s * 32 + kb);
            qf[s].u[1] = *(const uint4*)(qp + s * 32 + kb + 16);
        }
    }

    v8f oacc[4];
    float mrun[8], lrun[8];
    #pragma unroll
    for (int j = 0; j < 4; ++j)
        #pragma unroll
        for (int e = 0; e < 8; ++e) oacc[j][e] = 0.0f;
    #pragma unroll
    for (int r = 0; r < 8; ++r) { mrun[r] = -3.0e38f; lrun[r] = 0.0f; }

    const int* maskb = mask + (size_t)b * L_ * L_;

    for (int kb0 = 0; kb0 < L_; kb0 += 64) {
        // Async-stage K block [64 key][64 d] and V block [64 d][64 key]
        // (512 contiguous 16B chunks each, 2 per thread per tile).
        {
            int c0 = tid, c1 = tid + 256;
            int r0 = c0 >> 3, d0 = (c0 & 7) << 3;
            int r1 = c1 >> 3, d1 = (c1 & 7) << 3;
            async_ld16(&Klds[r0 * 64 + d0], &Km[head + (size_t)(kb0 + r0) * HD_ + d0]);
            async_ld16(&Klds[r1 * 64 + d1], &Km[head + (size_t)(kb0 + r1) * HD_ + d1]);
            async_ld16(&Vlds[r0 * 64 + d0], &Vt[vhead + ((size_t)r0 << 11) + kb0 + d0]);
            async_ld16(&Vlds[r1 * 64 + d1], &Vt[vhead + ((size_t)r1 << 11) + kb0 + d1]);
        }
        // Prefetch next mask block rows while DMA runs
        if (kb0 + 64 < L_)
            __builtin_prefetch(&maskb[(size_t)(qblk * 128 + (tid & 127)) * L_ +
                                      kb0 + 64 + ((tid >> 7) << 5)], 0, 0);
        wait_async0();
        __syncthreads();

        // Scores: S[16 x 64] = Q_tile @ K_block^T  (4 n-tiles x 2 k-steps)
        v8f sc[4];
        #pragma unroll
        for (int nt = 0; nt < 4; ++nt) {
            #pragma unroll
            for (int e = 0; e < 8; ++e) sc[nt][e] = 0.0f;
            #pragma unroll
            for (int s = 0; s < 2; ++s) {
                Frag16 bfr;
                const int key = nt * 16 + row16;
                const int kgd = s * 32 + (hi << 3);
                bfr.u[0] = *(const uint4*)&Klds[key * 64 + kgd];
                bfr.u[1] = *(const uint4*)&Klds[key * 64 + kgd + 16];
                sc[nt] = __builtin_amdgcn_wmma_f32_16x16x32_bf16(
                    false, qf[s].v, false, bfr.v, (short)0, sc[nt], false, false);
            }
        }

        // Mask + scale + online softmax (per C-element row r)
        const int kcol = kb0 + row16;
        #pragma unroll
        for (int r = 0; r < 8; ++r) {
            const int qrow = q_base + r + hi * 8;
            const int* mrow = maskb + (size_t)qrow * L_;
            float s0[4];
            float rmax = -3.0e38f;
            #pragma unroll
            for (int nt = 0; nt < 4; ++nt) {
                const int mv = mrow[kcol + nt * 16];
                const float e = mv ? sc[nt][r] * 0.125f : -1.25e19f;  // (-1e20)/8
                s0[nt] = e;
                rmax = fmaxf(rmax, e);
            }
            #pragma unroll
            for (int off = 1; off < 16; off <<= 1)
                rmax = fmaxf(rmax, __shfl_xor(rmax, off, 32));
            const float mnew  = fmaxf(mrun[r], rmax);
            const float scale = __expf(mrun[r] - mnew);
            float rsum = 0.0f;
            #pragma unroll
            for (int nt = 0; nt < 4; ++nt) {
                const float p = __expf(s0[nt] - mnew);
                rsum += p;
                Plds[wave][(r + hi * 8) * 64 + nt * 16 + row16] = (__bf16)p;
            }
            #pragma unroll
            for (int off = 1; off < 16; off <<= 1)
                rsum += __shfl_xor(rsum, off, 32);
            lrun[r] = lrun[r] * scale + rsum;
            mrun[r] = mnew;
            #pragma unroll
            for (int j = 0; j < 4; ++j) oacc[j][r] *= scale;
        }

        // O += P @ V_block  (A = P from per-wave LDS, B = Vlds[d][key])
        Frag16 pf[2];
        #pragma unroll
        for (int s = 0; s < 2; ++s) {
            pf[s].u[0] = *(const uint4*)&Plds[wave][row16 * 64 + s * 32 + kb];
            pf[s].u[1] = *(const uint4*)&Plds[wave][row16 * 64 + s * 32 + kb + 16];
        }
        #pragma unroll
        for (int j = 0; j < 4; ++j) {
            #pragma unroll
            for (int s = 0; s < 2; ++s) {
                Frag16 bfr;
                const int dn  = j * 16 + row16;
                const int kgd = s * 32 + (hi << 3);
                bfr.u[0] = *(const uint4*)&Vlds[dn * 64 + kgd];
                bfr.u[1] = *(const uint4*)&Vlds[dn * 64 + kgd + 16];
                oacc[j] = __builtin_amdgcn_wmma_f32_16x16x32_bf16(
                    false, pf[s].v, false, bfr.v, (short)0, oacc[j], false, false);
            }
        }
        __syncthreads();
    }

    // Normalize and write O (bf16, row-major [B*L, 1024], head-interleaved cols)
    #pragma unroll
    for (int j = 0; j < 4; ++j) {
        #pragma unroll
        for (int r = 0; r < 8; ++r) {
            const int qrow = q_base + r + hi * 8;
            const float val = oacc[j][r] / lrun[r];
            O[(size_t)(b * L_ + qrow) * EMBED_ + h * 64 + j * 16 + row16] = (__bf16)val;
        }
    }
}

// ---------------------------------------------------------------------------
// Launch
// ---------------------------------------------------------------------------
extern "C" void kernel_launch(void* const* d_in, const int* in_sizes, int n_in,
                              void* d_out, int out_size, void* d_ws, size_t ws_size,
                              hipStream_t stream) {
    const float* x    = (const float*)d_in[0];
    const int*   mask = (const int*)d_in[1];
    const float* Wq   = (const float*)d_in[2];
    const float* bq   = (const float*)d_in[3];
    const float* Wk   = (const float*)d_in[4];
    const float* bk   = (const float*)d_in[5];
    const float* Wv   = (const float*)d_in[6];
    const float* bv   = (const float*)d_in[7];
    const float* Wo   = (const float*)d_in[8];
    const float* bo   = (const float*)d_in[9];
    float* out = (float*)d_out;

    char* ws = (char*)d_ws;
    const size_t XN = (size_t)M_ * EMBED_;       // 8M elements
    const size_t WN = (size_t)EMBED_ * EMBED_;   // 1M elements
    __bf16* xb  = (__bf16*)ws;
    __bf16* wqt = xb  + XN;
    __bf16* wkt = wqt + WN;
    __bf16* wvt = wkt + WN;
    __bf16* wot = wvt + WN;
    __bf16* Qb  = wot + WN;
    __bf16* Kb  = Qb  + XN;
    __bf16* Vb  = Kb  + XN;   // [B,H,D,L]
    __bf16* Ob  = Vb  + XN;

    cvt_f32_bf16<<<(int)((XN + 255) / 256), 256, 0, stream>>>(x, xb, (int)XN);
    cvt_w_transpose<<<(int)(WN / 256), 256, 0, stream>>>(Wq, wqt);
    cvt_w_transpose<<<(int)(WN / 256), 256, 0, stream>>>(Wk, wkt);
    cvt_w_transpose<<<(int)(WN / 256), 256, 0, stream>>>(Wv, wvt);
    cvt_w_transpose<<<(int)(WN / 256), 256, 0, stream>>>(Wo, wot);

    dim3 g(M_ / 128, EMBED_ / 128);
    gemm_bf16<<<g, 256, 0, stream>>>(xb, wqt, bq, (void*)Qb, 0);
    gemm_bf16<<<g, 256, 0, stream>>>(xb, wkt, bk, (void*)Kb, 0);
    gemm_bf16<<<g, 256, 0, stream>>>(xb, wvt, bv, (void*)Vb, 2);

    attn_kernel<<<B_ * HEADS_ * (L_ / 128), 256, 0, stream>>>(Qb, Kb, Vb, mask, Ob);

    gemm_bf16<<<g, 256, 0, stream>>>(Ob, wot, bo, (void*)out, 1);
}
